// Head_50861002719698
// MI455X (gfx1250) — matrix operand order repaired and checked by reading, not compiled, not run on hardware
//
#include <hip/hip_runtime.h>

// ---------------------------------------------------------------------------
// Single-head attention (no mask), B=32, S=2048, n_embd = head_size = 64.
//   q = x Wq^T ; k = x Wk^T ; v = x Wv^T        (f32 in -> f16 staged)
//   out = softmax(q k^T / 8) v                  (f32 accum via WMMA)
// Kernel A: WMMA projections, LDS-staged coalesced stores.
// Kernel B: flash attention, ping-pong LDS K/V, 1 barrier/chunk, WMMA.
// ---------------------------------------------------------------------------

typedef __attribute__((ext_vector_type(16))) _Float16 v16h;
typedef __attribute__((ext_vector_type(8)))  _Float16 v8h;
typedef __attribute__((ext_vector_type(8)))  float    v8f;

#define BATCH 32
#define SEQ   2048
#define DIM   64
#define SCORE_SCALE 0.125f   // 1/sqrt(64)

static __device__ __forceinline__ v16h cat8(const _Float16* p0, const _Float16* p1) {
    v8h a = *(const v8h*)p0;
    v8h b = *(const v8h*)p1;
    v16h r;
#pragma unroll
    for (int i = 0; i < 8; ++i) { r[i] = a[i]; r[8 + i] = b[i]; }
    return r;
}

// Intra-wave LDS producer->consumer fence: DS ops of one wave retire in order;
// we only need the writes visible before cross-lane reads, and to pin the
// compiler's ordering around the exchange.
static __device__ __forceinline__ void wave_lds_fence() {
    __builtin_amdgcn_wave_barrier();
    asm volatile("s_wait_dscnt 0x0" ::: "memory");
    __builtin_amdgcn_wave_barrier();
}

// ===========================================================================
// Kernel A: q/k/v projections. One wave per 16-row tile of x.
//   q,k stored f16 row-major [B*S][64]; v stored f16 transposed [B][64][S].
//   D tiles staged in LDS, then stored coalesced (b128).
// ===========================================================================
__global__ __launch_bounds__(256)
void qkv_proj_kernel(const float* __restrict__ x,
                     const float* __restrict__ Wq,
                     const float* __restrict__ Wk,
                     const float* __restrict__ Wv,
                     _Float16* __restrict__ qws,
                     _Float16* __restrict__ kws,
                     _Float16* __restrict__ vTws)
{
    __shared__ _Float16 sbuf[8][16][80];   // 16x64 tile + pad (160B row stride)

    const int lane = threadIdx.x & 31;
    const int wave = threadIdx.x >> 5;
    const int tile = blockIdx.x * 8 + wave;      // 16-row tile, 4096 total
    const int row0 = tile * 16;                  // flat row = b*S + s
    const int m    = lane & 15;
    const bool hi  = lane >= 16;
    const int k8   = hi ? 8 : 0;
    const int k16  = hi ? 16 : 0;
    const int hoff = hi ? 8 : 0;

    // ---- A fragments of x rows (M=rows, K=embed), f32 -> f16 ----
    const float* xr = x + (size_t)(row0 + m) * DIM;
    v16h a_lo, a_hi;
#pragma unroll
    for (int i = 0; i < 8; ++i) {
        a_lo[i]     = (_Float16)xr[k8 + i];
        a_lo[8 + i] = (_Float16)xr[16 + k8 + i];
        a_hi[i]     = (_Float16)xr[32 + k8 + i];
        a_hi[8 + i] = (_Float16)xr[48 + k8 + i];
    }

    const int b  = row0 / SEQ;
    const int s0 = row0 % SEQ;
    const float* Ws[3] = { Wq, Wk, Wv };

#pragma unroll
    for (int w = 0; w < 3; ++w) {
        const float* W = Ws[w];
#pragma unroll
        for (int nb = 0; nb < 4; ++nb) {
            // B = W^T (K=e, N=h): lane column h = nb*16 + m, contiguous in W row.
            const float* wr = W + (size_t)(nb * 16 + m) * DIM;
            v16h b_lo, b_hi;
#pragma unroll
            for (int i = 0; i < 16; ++i) {
                b_lo[i] = (_Float16)wr[k8 + (i & 7) + ((i >> 3) << 4)];
                b_hi[i] = (_Float16)wr[32 + k8 + (i & 7) + ((i >> 3) << 4)];
            }
            v8f c = {};
            c = __builtin_amdgcn_wmma_f32_16x16x32_f16(false, a_lo, false, b_lo,
                                                       (short)0, c, false, false);
            c = __builtin_amdgcn_wmma_f32_16x16x32_f16(false, a_hi, false, b_hi,
                                                       (short)0, c, false, false);
            // D layout: col n = m; row = r + hoff  -> stage in LDS
#pragma unroll
            for (int r = 0; r < 8; ++r)
                sbuf[wave][r + hoff][nb * 16 + m] = (_Float16)c[r];
        }
        wave_lds_fence();

        if (w < 2) {
            // q/k: row-major coalesced stores, 2x 32B per lane
            _Float16* dst = (w == 0 ? qws : kws) + (size_t)row0 * DIM;
#pragma unroll
            for (int t = 0; t < 2; ++t) {
                const int seg  = lane + t * 32;        // 64 segs of 16 halfs
                const int row  = seg >> 2;
                const int part = (seg & 3) * 16;
                *(v16h*)(dst + (size_t)row * DIM + part) =
                    *(const v16h*)(&sbuf[wave][row][part]);
            }
        } else {
            // v: transposed [B][64][S]; per lane two head-rows of 16 halfs
#pragma unroll
            for (int t = 0; t < 2; ++t) {
                const int h = lane + t * 32;
                v16h col;
#pragma unroll
                for (int i = 0; i < 16; ++i) col[i] = sbuf[wave][i][h];
                *(v16h*)(vTws + ((size_t)b * DIM + h) * SEQ + s0) = col;
            }
        }
        wave_lds_fence();   // before next w overwrites sbuf
    }
}

// ===========================================================================
// Kernel B: flash attention. Block = 4 waves = 64 query rows of one batch.
// Key loop in chunks of 32; ping-pong K/V^T LDS buffers; register prefetch;
// one workgroup barrier per chunk; wave-private softmax exchange.
// ===========================================================================
__global__ __launch_bounds__(128)
void attn_kernel(const _Float16* __restrict__ qws,
                 const _Float16* __restrict__ kws,
                 const _Float16* __restrict__ vTws,
                 float* __restrict__ out)
{
    __shared__ _Float16 kbuf[2][32][64];    // [buf][key][head]       8 KB
    __shared__ _Float16 vbuf[2][64][32];    // [buf][head][key] (V^T) 8 KB
    __shared__ float    stile[4][16][32];   // per-wave scores        8 KB
    __shared__ _Float16 ptile[4][16][32];   // per-wave P (f16)       4 KB
    __shared__ float    salpha[4][16];
    __shared__ float    slsum[4][16];

    const int tid  = threadIdx.x;
    const int lane = tid & 31;
    const int wave = tid >> 5;
    const int b    = blockIdx.x >> 5;           // 32 q-tiles per batch
    const int qw   = (blockIdx.x & 31) * 64 + wave * 16;
    const int m    = lane & 15;
    const bool hi  = lane >= 16;
    const int k8   = hi ? 8 : 0;
    const int k16  = hi ? 16 : 0;
    const int hoff = hi ? 8 : 0;

    // ---- Q A-fragments (M=query rows, K=head dim 64 -> two steps) ----
    const _Float16* qr = qws + ((size_t)b * SEQ + qw + m) * DIM;
    const v16h aq_lo = cat8(qr + k16,      qr + 8 + k16);
    const v16h aq_hi = cat8(qr + 32 + k16, qr + 40 + k16);

    v8f O[4];
#pragma unroll
    for (int t = 0; t < 4; ++t) O[t] = (v8f){};
    float m_i = -INFINITY, l_i = 0.0f;          // lanes 0..15: row = lane

    const _Float16* ksrc = kws  + ((size_t)b * SEQ) * DIM;
    const _Float16* vsrc = vTws + ((size_t)b * DIM) * SEQ;

    // per-thread staging coordinates
    const int hh = tid >> 1;                    // V^T head row
    const int pp = (tid & 1) * 16;              // V^T key part

    // ---- stage chunk 0 ----
    *(v16h*)(&kbuf[0][0][0] + tid * 16) = *(const v16h*)(ksrc + tid * 16);
    *(v16h*)(&vbuf[0][hh][pp])          = *(const v16h*)(vsrc + (size_t)hh * SEQ + pp);
    __syncthreads();

    for (int jc = 0; jc < SEQ / 32; ++jc) {
        const int cur = jc & 1;
        const int nxt = cur ^ 1;
        const int j0n = (jc + 1) * 32;

        // ---- register prefetch of next chunk (overlaps with WMMA below) ----
        v16h kreg = {}, vreg = {};
        if (jc < SEQ / 32 - 1) {
            kreg = *(const v16h*)(ksrc + (size_t)j0n * DIM + tid * 16);
            vreg = *(const v16h*)(vsrc + (size_t)hh * SEQ + j0n + pp);
            if (jc < SEQ / 32 - 2) {   // L2-latency hiding for chunk jc+2
                __builtin_prefetch(ksrc + (size_t)(j0n + 32) * DIM + tid * 16, 0, 1);
                __builtin_prefetch(vsrc + (size_t)hh * SEQ + j0n + 32 + pp, 0, 1);
            }
        }

        // ---- scores: two 16-key halves, K = head dim (2 steps of 32) ----
#pragma unroll
        for (int h = 0; h < 2; ++h) {
            const _Float16* kr = &kbuf[cur][16 * h + m][0];
            const v16h bl = *(const v16h*)(kr + k16);
            const v16h bh = *(const v16h*)(kr + 32 + k16);
            v8f c = {};
            c = __builtin_amdgcn_wmma_f32_16x16x32_f16(false, aq_lo, false, bl,
                                                       (short)0, c, false, false);
            c = __builtin_amdgcn_wmma_f32_16x16x32_f16(false, aq_hi, false, bh,
                                                       (short)0, c, false, false);
#pragma unroll
            for (int r = 0; r < 8; ++r)
                stile[wave][r + hoff][16 * h + m] = c[r] * SCORE_SCALE;
        }
        wave_lds_fence();

        // ---- online softmax: lane i (<16) owns query row i of the wave ----
        if (lane < 16) {
            float mx = m_i;
#pragma unroll
            for (int j = 0; j < 32; ++j) mx = fmaxf(mx, stile[wave][lane][j]);
            const float alpha = __expf(m_i - mx);
            float lsum = 0.0f;
#pragma unroll
            for (int j = 0; j < 32; ++j) {
                const float p = __expf(stile[wave][lane][j] - mx);
                lsum += p;
                ptile[wave][lane][j] = (_Float16)p;   // row-major == A-frag layout
            }
            l_i = l_i * alpha + lsum;
            m_i = mx;
            salpha[wave][lane] = alpha;
        }
        wave_lds_fence();

        // ---- rescale O by alpha(row), then O += P(16x32) x V(32x64) ----
        float ar[8];
#pragma unroll
        for (int r = 0; r < 8; ++r) ar[r] = salpha[wave][r + hoff];
#pragma unroll
        for (int t = 0; t < 4; ++t)
#pragma unroll
            for (int r = 0; r < 8; ++r) O[t][r] *= ar[r];

        const _Float16* pr = &ptile[wave][m][0];
        const v16h ap = cat8(pr + k16, pr + 8 + k16);
#pragma unroll
        for (int t = 0; t < 4; ++t) {
            const v16h bv = *(const v16h*)(&vbuf[cur][t * 16 + m][k16]);
            O[t] = __builtin_amdgcn_wmma_f32_16x16x32_f16(false, ap, false, bv,
                                                          (short)0, O[t], false, false);
        }

        // ---- commit prefetched chunk to the other buffer ----
        if (jc < SEQ / 32 - 1) {
            *(v16h*)(&kbuf[nxt][0][0] + tid * 16) = kreg;
            *(v16h*)(&vbuf[nxt][hh][pp])          = vreg;
        }
        __syncthreads();   // single workgroup barrier per chunk
    }

    // ---- finalize: divide by row sums, store f32 ----
    if (lane < 16) slsum[wave][lane] = l_i;
    wave_lds_fence();
    float lr[8];
#pragma unroll
    for (int r = 0; r < 8; ++r) lr[r] = 1.0f / slsum[wave][r + hoff];
#pragma unroll
    for (int t = 0; t < 4; ++t)
#pragma unroll
        for (int r = 0; r < 8; ++r)
            out[((size_t)b * SEQ + qw + r + hoff) * DIM + t * 16 + m] =
                O[t][r] * lr[r];
}

// ===========================================================================
extern "C" void kernel_launch(void* const* d_in, const int* in_sizes, int n_in,
                              void* d_out, int out_size, void* d_ws, size_t ws_size,
                              hipStream_t stream) {
    (void)in_sizes; (void)n_in; (void)out_size; (void)ws_size;
    const float* x  = (const float*)d_in[0];
    const float* Wq = (const float*)d_in[1];
    const float* Wk = (const float*)d_in[2];
    const float* Wv = (const float*)d_in[3];
    float* out = (float*)d_out;

    const size_t QK = (size_t)BATCH * SEQ * DIM;     // 4Mi f16 elements each
    _Float16* qws = (_Float16*)d_ws;
    _Float16* kws = qws + QK;
    _Float16* vT  = kws + QK;                        // total 24 MB of d_ws

    // 4096 16-row tiles / 8 waves per block
    qkv_proj_kernel<<<dim3((BATCH * SEQ / 16) / 8), dim3(256), 0, stream>>>(
        x, Wq, Wk, Wv, qws, kws, vT);
    // one block per (batch, 64-query tile)
    attn_kernel<<<dim3(BATCH * (SEQ / 64)), dim3(128), 0, stream>>>(
        qws, kws, vT, out);
}
